// gcn_self_attn_20023137534034
// MI455X (gfx1250) — matrix operand order repaired and checked by reading, not compiled, not run on hardware
//
#include <hip/hip_runtime.h>

// ---------------------------------------------------------------------------
// Types for CDNA5 WMMA
// ---------------------------------------------------------------------------
typedef __attribute__((ext_vector_type(16))) __bf16 v16bf;
typedef __attribute__((ext_vector_type(8)))  float  v8f;
typedef __attribute__((ext_vector_type(4)))  unsigned int v4u;
typedef __attribute__((ext_vector_type(8)))  int    v8i;
typedef __attribute__((ext_vector_type(4)))  int    v4i;

__device__ __forceinline__ unsigned short f2bfbits(float f) {
    unsigned u = __float_as_uint(f);
    u += 0x7FFFu + ((u >> 16) & 1u);          // round-to-nearest-even
    return (unsigned short)(u >> 16);
}
__device__ __forceinline__ float bfl(unsigned u) { return __uint_as_float(u << 16); }
__device__ __forceinline__ float bfh(unsigned u) { return __uint_as_float(u & 0xffff0000u); }

// ---------------------------------------------------------------------------
// K0: M = (Wq @ Wk^T) / sqrt(HID)   (2x2 core of the rank-2 attention)
// ---------------------------------------------------------------------------
__global__ void __launch_bounds__(256)
mscale_kernel(const float* __restrict__ Wq, const float* __restrict__ Wk,
              float* __restrict__ M, int HID) {
    float p0 = 0.f, p1 = 0.f, p2 = 0.f, p3 = 0.f;
    for (int d = threadIdx.x; d < HID; d += 256) {
        float q0 = Wq[d], q1 = Wq[HID + d];
        float k0 = Wk[d], k1 = Wk[HID + d];
        p0 = fmaf(q0, k0, p0); p1 = fmaf(q0, k1, p1);
        p2 = fmaf(q1, k0, p2); p3 = fmaf(q1, k1, p3);
    }
    for (int off = 16; off; off >>= 1) {
        p0 += __shfl_xor(p0, off, 32); p1 += __shfl_xor(p1, off, 32);
        p2 += __shfl_xor(p2, off, 32); p3 += __shfl_xor(p3, off, 32);
    }
    __shared__ float red[8][4];
    int wave = threadIdx.x >> 5, lane = threadIdx.x & 31;
    if (lane == 0) { red[wave][0]=p0; red[wave][1]=p1; red[wave][2]=p2; red[wave][3]=p3; }
    __syncthreads();
    if (threadIdx.x < 4) {
        float s = 0.f;
        for (int w = 0; w < 8; ++w) s += red[w][threadIdx.x];
        M[threadIdx.x] = s * rsqrtf((float)HID);
    }
}

// ---------------------------------------------------------------------------
// K1: rank-2 softmax attention.  y[i] = sum_j softmax_j(x_i M x_j^T) * x_j
// One block = 256 query nodes of one graph; graph's x tile lives in LDS.
// ---------------------------------------------------------------------------
__global__ void __launch_bounds__(256)
attn_kernel(const float* __restrict__ x, const float* __restrict__ M,
            float* __restrict__ y, int NPG) {
    __shared__ float2 sx[1024];                 // NPG <= 1024 for this workload
    const int bpg   = NPG >> 8;                 // blocks per graph (256 q-rows each)
    const int g     = blockIdx.x / bpg;
    const int local = (blockIdx.x % bpg) * 256 + threadIdx.x;
    const float2* xg = (const float2*)x + (size_t)g * NPG;
    for (int i = threadIdx.x; i < NPG; i += 256) sx[i] = xg[i];
    __syncthreads();
    const float m00 = M[0], m01 = M[1], m10 = M[2], m11 = M[3];
    float2 xi = sx[local];
    float a = fmaf(xi.x, m00, xi.y * m10);
    float b = fmaf(xi.x, m01, xi.y * m11);
    float mx = -3.4e38f;
    for (int j = 0; j < NPG; ++j) {
        float2 xj = sx[j];
        mx = fmaxf(mx, fmaf(a, xj.x, b * xj.y));
    }
    float ssum = 0.f, y0 = 0.f, y1 = 0.f;
    for (int j = 0; j < NPG; ++j) {
        float2 xj = sx[j];
        float p = __expf(fmaf(a, xj.x, b * xj.y) - mx);
        ssum += p; y0 = fmaf(p, xj.x, y0); y1 = fmaf(p, xj.y, y1);
    }
    float inv = 1.0f / ssum;
    ((float2*)y)[(size_t)g * NPG + local] = make_float2(y0 * inv, y1 * inv);
}

// ---------------------------------------------------------------------------
// K2: moments of y (for collapsed BatchNorm): sums of y0,y1,y0^2,y0y1,y1^2
// ---------------------------------------------------------------------------
__global__ void __launch_bounds__(256)
moments_kernel(const float2* __restrict__ y, float* __restrict__ mom, int N) {
    int i0 = blockIdx.x * blockDim.x + threadIdx.x;
    int stride = gridDim.x * blockDim.x;
    float s0=0.f, s1=0.f, s2=0.f, s3=0.f, s4=0.f;
    for (int i = i0; i < N; i += stride) {
        float2 v = y[i];
        s0 += v.x; s1 += v.y;
        s2 = fmaf(v.x, v.x, s2); s3 = fmaf(v.x, v.y, s3); s4 = fmaf(v.y, v.y, s4);
    }
    for (int off = 16; off; off >>= 1) {
        s0 += __shfl_xor(s0, off, 32); s1 += __shfl_xor(s1, off, 32);
        s2 += __shfl_xor(s2, off, 32); s3 += __shfl_xor(s3, off, 32);
        s4 += __shfl_xor(s4, off, 32);
    }
    if ((threadIdx.x & 31) == 0) {
        atomicAdd(&mom[0], s0); atomicAdd(&mom[1], s1); atomicAdd(&mom[2], s2);
        atomicAdd(&mom[3], s3); atomicAdd(&mom[4], s4);
    }
}

// ---------------------------------------------------------------------------
// K2b: fold BN into affine (A0,A1,C); pack Wm to bf16 pre-swizzled in WMMA
// B-fragment order [chunk][frag][lane][16 bf16]; fold all 32-dim output
// projections into combined vectors + scalar bias.  Single block, 512 threads.
// ---------------------------------------------------------------------------
__global__ void __launch_bounds__(512)
prep_kernel(const float* __restrict__ mom, const float* __restrict__ Wv,
            const float* __restrict__ gamma, const float* __restrict__ beta,
            const float* __restrict__ Wm,
            const float* __restrict__ Wnp, const float* __restrict__ bnp,
            const float* __restrict__ Wb_root, const float* __restrict__ Wb_msg,
            const float* __restrict__ bb,
            const float* __restrict__ Wf_root, const float* __restrict__ Wf_msg,
            const float* __restrict__ bf_,
            const float* __restrict__ Wpb, const float* __restrict__ bpb,
            const float* __restrict__ Wpf, const float* __restrict__ bpf,
            float* __restrict__ A0, float* __restrict__ A1, float* __restrict__ C,
            unsigned short* __restrict__ WmBf, float* __restrict__ cvec,
            int N, int HID) {
    const int d = threadIdx.x;
    const float invN = 1.0f / (float)N;
    const float ym0 = mom[0]*invN, ym1 = mom[1]*invN;
    const float c00 = mom[2]*invN - ym0*ym0;
    const float c01 = mom[3]*invN - ym0*ym1;
    const float c11 = mom[4]*invN - ym1*ym1;
    if (d < HID) {
        float wv0 = Wv[d], wv1 = Wv[HID + d];
        float mu  = fmaf(ym0, wv0, ym1 * wv1);
        float var = c00*wv0*wv0 + 2.f*c01*wv0*wv1 + c11*wv1*wv1;
        float g   = gamma[d] * rsqrtf(var + 1e-5f);
        A0[d] = g * wv0; A1[d] = g * wv1; C[d] = beta[d] - g * mu;
    }
    // Pre-swizzle Wm into B-fragment order: triple t = (kc, frag, lane).
    // For (kc,f,l): l16=l>>4, lm=l&15, n=lm+16*f, kB=kc*32+l16*16,
    //   elems j=0..7: (Wm[(kB+2j)*32+n], Wm[(kB+2j+1)*32+n])  -> 16 bf16 at t*16.
    const int triples = (HID >> 5) * 64;        // chunks * 2 frags * 32 lanes
    for (int t = d; t < triples; t += 512) {
        const int kc = t >> 6, rem = t & 63;
        const int f  = rem >> 5, l = rem & 31;
        const int l16 = l >> 4, lm = l & 15;
        const int n  = lm + 16 * f;
        const int kB = kc * 32 + l16 * 16;
        unsigned short* dst = WmBf + (size_t)t * 16;
        #pragma unroll
        for (int j = 0; j < 8; ++j) {
            dst[2*j]   = f2bfbits(Wm[(kB + 2*j)     * 32 + n]);
            dst[2*j+1] = f2bfbits(Wm[(kB + 2*j + 1) * 32 + n]);
        }
    }
    if (d < 32) {
        float wc = Wnp[d], vb = 0.f, vf = 0.f;
        for (int j = 0; j < 32; ++j) {
            wc = fmaf(Wb_root[d*32 + j], Wpb[j], wc);
            wc = fmaf(Wf_root[d*32 + j], Wpf[j], wc);
            vb = fmaf(Wb_msg [d*32 + j], Wpb[j], vb);
            vf = fmaf(Wf_msg [d*32 + j], Wpf[j], vf);
        }
        cvec[d] = wc; cvec[32 + d] = vb; cvec[64 + d] = vf;
    }
    if (d == 0) {
        float c0 = bnp[0] + bpb[0] + bpf[0];
        for (int j = 0; j < 32; ++j) { c0 = fmaf(bb[j], Wpb[j], c0); c0 = fmaf(bf_[j], Wpf[j], c0); }
        cvec[96] = c0;
    }
}

// ---------------------------------------------------------------------------
// K3: feat = PReLU(BN(y@Wv)) @ Wm + bm  via v_wmma_f32_16x16x32_bf16.
// One wave owns a 16-node tile; h is regenerated on the fly from (y0,y1) and
// the folded affine. Wm arrives in LDS pre-swizzled (TDM DMA), so each B
// fragment is a single 32-byte vector load per lane.
// ---------------------------------------------------------------------------
__global__ void __launch_bounds__(256)
feat_wmma_kernel(const float2* __restrict__ y,
                 const float* __restrict__ A0g, const float* __restrict__ A1g,
                 const float* __restrict__ Cg,
                 const unsigned short* __restrict__ WmBf,
                 const float* __restrict__ bm, const float* __restrict__ prelu_a,
                 float* __restrict__ feat, int N, int HID) {
    __shared__ float sA0[512], sA1[512], sC[512];
    __shared__ unsigned short sWm[512 * 32];     // pre-swizzled B fragments, 32 KB
    const int tid = threadIdx.x;
    for (int i = tid; i < HID; i += 256) { sA0[i]=A0g[i]; sA1[i]=A1g[i]; sC[i]=Cg[i]; }

#if defined(__has_builtin) && __has_builtin(__builtin_amdgcn_tensor_load_to_lds) && defined(__gfx1250__)
    // Tensor Data Mover: DMA HID*32 bf16 elements (one contiguous row) into LDS.
    // 6-arg form on this toolchain: (g0 v4u, g1 v8i, g2 v4i, g3 v4i, g4 v8i, cpol).
    if ((tid >> 5) == 0) {                        // wave 0 only (uniform branch)
        unsigned long long ga = (unsigned long long)(const void*)WmBf;
        unsigned lds_off = (unsigned)(unsigned long long)(void*)&sWm[0]; // low32 of flat = LDS offset
        unsigned elems = (unsigned)HID * 32u;
        v4u g0 = { 1u,                                    // count=1, user mode
                   lds_off,
                   (unsigned)(ga & 0xffffffffu),
                   (unsigned)(((ga >> 32) & 0x01FFFFFFu) | 0x80000000u) }; // addr[56:32] | type=2
        v8i g1 = { (int)0x00010000u,                      // wg_mask=0, data_size=2B
                   (int)((elems & 0xFFFFu) << 16),        // tensor_dim0 lo16
                   (int)((elems >> 16) | (1u << 16)),     // tensor_dim0 hi16 | tensor_dim1=1
                   (int)((elems & 0xFFFFu) << 16),        // tile_dim0
                   (int)1,                                // tile_dim1=1, tile_dim2=0
                   (int)elems,                            // tensor_dim0_stride lo32
                   0, 0 };
        v4i z4 = { 0, 0, 0, 0 };
        v8i z8 = { 0, 0, 0, 0, 0, 0, 0, 0 };
        __builtin_amdgcn_tensor_load_to_lds(g0, g1, z4, z4, z8, 0);
        __builtin_amdgcn_s_wait_tensorcnt(0);
    }
#else
    {
        const unsigned int* src = (const unsigned int*)WmBf;
        unsigned int* dst = (unsigned int*)sWm;
        for (int i = tid; i < HID * 16; i += 256) dst[i] = src[i];
    }
#endif
    __syncthreads();

    const float pa   = *prelu_a;
    const int wave   = tid >> 5, lane = tid & 31;
    const int l16    = lane >> 4, lm = lane & 15;
    const int nodeBase = blockIdx.x * 128 + wave * 16;
    const float2 yv  = y[nodeBase + lm];
    const v16bf* sWmV = (const v16bf*)sWm;       // fragment-granular view

    v8f acc0 = {0,0,0,0,0,0,0,0};
    v8f acc1 = {0,0,0,0,0,0,0,0};
    const int chunks = HID >> 5;
    for (int kc = 0; kc < chunks; ++kc) {
        // A fragment: rows = 16 nodes, K chunk of 32 (bf16 A-layout, 16x32)
        const int kA = kc*32 + l16*16;
        v16bf av;
        #pragma unroll
        for (int j = 0; j < 8; ++j) {
            int k0 = kA + 2*j;
            float h0 = fmaf(sA0[k0],   yv.x, fmaf(sA1[k0],   yv.y, sC[k0]));
            float h1 = fmaf(sA0[k0+1], yv.x, fmaf(sA1[k0+1], yv.y, sC[k0+1]));
            h0 = h0 >= 0.f ? h0 : pa * h0;
            h1 = h1 >= 0.f ? h1 : pa * h1;
            av[2*j] = (__bf16)h0;                // native v_cvt f32->bf16
            av[2*j+1] = (__bf16)h1;
        }
        // B fragments: one 32-byte vector load per lane (pre-swizzled layout)
        v16bf bv0 = sWmV[(kc*2 + 0)*32 + lane];
        v16bf bv1 = sWmV[(kc*2 + 1)*32 + lane];
        acc0 = __builtin_amdgcn_wmma_f32_16x16x32_bf16(false, av, false, bv0,
                                                       (short)0, acc0, false, false);
        acc1 = __builtin_amdgcn_wmma_f32_16x16x32_bf16(false, av, false, bv1,
                                                       (short)0, acc1, false, false);
    }
    const float bml = bm[lm], bmh = bm[16 + lm];
    #pragma unroll
    for (int r = 0; r < 8; ++r) {
        int node = nodeBase + r + 8*l16;            // C/D layout: M = r + 8*(lane>>4)
        feat[(size_t)node*32 + lm]      = acc0[r] + bml;
        feat[(size_t)node*32 + 16 + lm] = acc1[r] + bmh;
    }
}

// ---------------------------------------------------------------------------
// K3b: per-node scalars from feat: base = feat.wcomb + c0, sb = feat.vbc, sf = feat.vfc
// ---------------------------------------------------------------------------
__global__ void __launch_bounds__(256)
node_vec_kernel(const float* __restrict__ feat, const float* __restrict__ cvec,
                float* __restrict__ base, float* __restrict__ sb,
                float* __restrict__ sf, int N) {
    __shared__ float sw[97];
    if (threadIdx.x < 97) sw[threadIdx.x] = cvec[threadIdx.x];
    __syncthreads();
    int n = blockIdx.x * blockDim.x + threadIdx.x;
    if (n >= N) return;
    const float* f = feat + (size_t)n * 32;
    float a = 0.f, b = 0.f, c = 0.f;
    #pragma unroll
    for (int k = 0; k < 32; ++k) {
        float fv = f[k];
        a = fmaf(fv, sw[k], a); b = fmaf(fv, sw[32+k], b); c = fmaf(fv, sw[64+k], c);
    }
    base[n] = a + sw[96]; sb[n] = b; sf[n] = c;
}

// ---------------------------------------------------------------------------
// K4a: row inverse norms (and optional bf16 repack of vis for L2 residency).
// One wave per node row.
// ---------------------------------------------------------------------------
__global__ void __launch_bounds__(256)
norm_convert_kernel(const float* __restrict__ vis, float* __restrict__ inv,
                    unsigned int* __restrict__ visbf /* nullable */, int D, int N) {
    int wid  = (blockIdx.x * blockDim.x + threadIdx.x) >> 5;
    int lane = threadIdx.x & 31;
    if (wid >= N) return;
    const float2* row = (const float2*)(vis + (size_t)wid * D);
    unsigned int* out = visbf ? visbf + ((size_t)wid * D >> 1) : nullptr;
    float acc = 0.f;
    const int pairs = D >> 1;
    for (int i = lane; i < pairs; i += 32) {
        float2 v = row[i];
        acc = fmaf(v.x, v.x, acc); acc = fmaf(v.y, v.y, acc);
        if (out) out[i] = (unsigned)f2bfbits(v.x) | ((unsigned)f2bfbits(v.y) << 16);
    }
    for (int off = 16; off; off >>= 1) acc += __shfl_xor(acc, off, 32);
    if (lane == 0) inv[wid] = rsqrtf(acc + 1e-8f);
}

// ---------------------------------------------------------------------------
// K4c: edge kernel (bf16 path). One wave per edge:
//   w = cos(vis[src],vis[dst]); num[dst] += w*s[src]; deg[dst] += 1
// ---------------------------------------------------------------------------
__global__ void __launch_bounds__(256)
edge_cos_bf16_kernel(const unsigned short* __restrict__ vis,
                     const float* __restrict__ inv, const int* __restrict__ edge,
                     const float* __restrict__ s, float* __restrict__ num,
                     float* __restrict__ deg, int E, int D) {
    int e    = (blockIdx.x * blockDim.x + threadIdx.x) >> 5;
    int lane = threadIdx.x & 31;
    if (e >= E) return;
    int src = edge[e], dst = edge[E + e];
    const uint2* a = (const uint2*)(vis + (size_t)src * D);
    const uint2* b = (const uint2*)(vis + (size_t)dst * D);
    __builtin_prefetch((const char*)b + lane * 128, 0, 0);   // global_prefetch_b8
    float acc = 0.f;
    const int q = D >> 2;
    for (int i = lane; i < q; i += 32) {
        uint2 ua = a[i], ub = b[i];
        acc = fmaf(bfl(ua.x), bfl(ub.x), acc); acc = fmaf(bfh(ua.x), bfh(ub.x), acc);
        acc = fmaf(bfl(ua.y), bfl(ub.y), acc); acc = fmaf(bfh(ua.y), bfh(ub.y), acc);
    }
    for (int off = 16; off; off >>= 1) acc += __shfl_xor(acc, off, 32);
    if (lane == 0) {
        float w = acc * inv[src] * inv[dst];
        atomicAdd(&num[dst], w * s[src]);
        atomicAdd(&deg[dst], 1.0f);
    }
}

// K4c fallback: f32 gathers (workspace too small for bf16 mirror)
__global__ void __launch_bounds__(256)
edge_cos_f32_kernel(const float* __restrict__ vis, const float* __restrict__ inv,
                    const int* __restrict__ edge, const float* __restrict__ s,
                    float* __restrict__ num, float* __restrict__ deg, int E, int D) {
    int e    = (blockIdx.x * blockDim.x + threadIdx.x) >> 5;
    int lane = threadIdx.x & 31;
    if (e >= E) return;
    int src = edge[e], dst = edge[E + e];
    const float4* a = (const float4*)(vis + (size_t)src * D);
    const float4* b = (const float4*)(vis + (size_t)dst * D);
    __builtin_prefetch((const char*)b + lane * 128, 0, 0);
    float acc = 0.f;
    const int q = D >> 2;
    for (int i = lane; i < q; i += 32) {
        float4 va = a[i], vb = b[i];
        acc = fmaf(va.x, vb.x, acc); acc = fmaf(va.y, vb.y, acc);
        acc = fmaf(va.z, vb.z, acc); acc = fmaf(va.w, vb.w, acc);
    }
    for (int off = 16; off; off >>= 1) acc += __shfl_xor(acc, off, 32);
    if (lane == 0) {
        float w = acc * inv[src] * inv[dst];
        atomicAdd(&num[dst], w * s[src]);
        atomicAdd(&deg[dst], 1.0f);
    }
}

// ---------------------------------------------------------------------------
// K6: out = base + numb/max(degb,1) + numf/max(degf,1)
// ---------------------------------------------------------------------------
__global__ void __launch_bounds__(256)
final_kernel(const float* __restrict__ base, const float* __restrict__ numb,
             const float* __restrict__ degb, const float* __restrict__ numf,
             const float* __restrict__ degf, float* __restrict__ out, int N) {
    int n = blockIdx.x * blockDim.x + threadIdx.x;
    if (n >= N) return;
    out[n] = base[n] + numb[n] / fmaxf(degb[n], 1.f) + numf[n] / fmaxf(degf[n], 1.f);
}

// ---------------------------------------------------------------------------
// Host launcher
// ---------------------------------------------------------------------------
extern "C" void kernel_launch(void* const* d_in, const int* in_sizes, int n_in,
                              void* d_out, int out_size, void* d_ws, size_t ws_size,
                              hipStream_t stream) {
    (void)n_in; (void)out_size;
    const float* x        = (const float*)d_in[0];
    const float* vis_body = (const float*)d_in[1];
    const float* vis_face = (const float*)d_in[2];
    const int*   e_body   = (const int*)d_in[3];
    const int*   e_face   = (const int*)d_in[4];
    const float* Wq = (const float*)d_in[6];
    const float* Wk = (const float*)d_in[7];
    const float* Wv = (const float*)d_in[8];
    const float* gamma = (const float*)d_in[9];
    const float* beta  = (const float*)d_in[10];
    const float* prelu = (const float*)d_in[11];
    const float* Wm = (const float*)d_in[12];
    const float* bm = (const float*)d_in[13];
    const float* Wnp = (const float*)d_in[14];
    const float* bnp = (const float*)d_in[15];
    const float* Wb_root = (const float*)d_in[16];
    const float* Wb_msg  = (const float*)d_in[17];
    const float* bb      = (const float*)d_in[18];
    const float* Wf_root = (const float*)d_in[19];
    const float* Wf_msg  = (const float*)d_in[20];
    const float* bf_     = (const float*)d_in[21];
    const float* Wpb = (const float*)d_in[22];
    const float* bpb = (const float*)d_in[23];
    const float* Wpf = (const float*)d_in[24];
    const float* bpf = (const float*)d_in[25];

    const int N   = in_sizes[0] / 2;
    const int VB  = in_sizes[1] / N;
    const int VF  = in_sizes[2] / N;
    const int E   = in_sizes[3] / 2;
    const int HID = in_sizes[6] / 2;
    const int B   = 32;                 // reference batch_size (device scalar unreadable in capture)
    const int NPG = N / B;

    char* ws = (char*)d_ws;
    size_t off = 0;
    auto alloc = [&](size_t bytes) -> size_t {
        off = (off + 255) & ~(size_t)255;
        size_t o = off; off += bytes; return o;
    };
    const size_t oM    = alloc(4 * 4);
    const size_t oMom  = alloc(8 * 4);
    const size_t oCvec = alloc(128 * 4);
    const size_t oA0   = alloc((size_t)HID * 4);
    const size_t oA1   = alloc((size_t)HID * 4);
    const size_t oC    = alloc((size_t)HID * 4);
    const size_t oWmBf = alloc((size_t)HID * 32 * 2);
    const size_t oY    = alloc((size_t)N * 2 * 4);
    const size_t oFeat = alloc((size_t)N * 32 * 4);
    const size_t oBase = alloc((size_t)N * 4);
    const size_t oSb   = alloc((size_t)N * 4);
    const size_t oSf   = alloc((size_t)N * 4);
    const size_t oNumb = alloc((size_t)N * 4);
    const size_t oDegb = alloc((size_t)N * 4);
    const size_t oNumf = alloc((size_t)N * 4);
    const size_t oDegf = alloc((size_t)N * 4);
    const size_t oInvb = alloc((size_t)N * 4);
    const size_t oInvf = alloc((size_t)N * 4);
    const size_t oVbfB = alloc((size_t)N * VB * 2);
    const size_t oVbfF = alloc((size_t)N * VF * 2);
    const bool useBf = (ws_size >= off);   // bf16 vis mirror fits -> L2-resident gathers

    // Zero accumulators every launch (graph replays do not re-poison).
    (void)hipMemsetAsync(ws + oMom,  0, 32, stream);
    (void)hipMemsetAsync(ws + oNumb, 0, (size_t)N * 4, stream);
    (void)hipMemsetAsync(ws + oDegb, 0, (size_t)N * 4, stream);
    (void)hipMemsetAsync(ws + oNumf, 0, (size_t)N * 4, stream);
    (void)hipMemsetAsync(ws + oDegf, 0, (size_t)N * 4, stream);

    float* pM    = (float*)(ws + oM);
    float* pMom  = (float*)(ws + oMom);
    float* pCvec = (float*)(ws + oCvec);
    float* pA0   = (float*)(ws + oA0);
    float* pA1   = (float*)(ws + oA1);
    float* pC    = (float*)(ws + oC);
    unsigned short* pWmBf = (unsigned short*)(ws + oWmBf);
    float* pY    = (float*)(ws + oY);
    float* pFeat = (float*)(ws + oFeat);
    float* pBase = (float*)(ws + oBase);
    float* pSb   = (float*)(ws + oSb);
    float* pSf   = (float*)(ws + oSf);
    float* pNumb = (float*)(ws + oNumb);
    float* pDegb = (float*)(ws + oDegb);
    float* pNumf = (float*)(ws + oNumf);
    float* pDegf = (float*)(ws + oDegf);
    float* pInvb = (float*)(ws + oInvb);
    float* pInvf = (float*)(ws + oInvf);
    unsigned int* pVbfB = (unsigned int*)(ws + oVbfB);
    unsigned int* pVbfF = (unsigned int*)(ws + oVbfF);

    // K0: 2x2 attention core
    mscale_kernel<<<1, 256, 0, stream>>>(Wq, Wk, pM, HID);
    // K1: rank-2 two-pass softmax attention -> y[N,2]
    attn_kernel<<<B * (NPG / 256), 256, 0, stream>>>(x, pM, pY, NPG);
    // K2: y moments for collapsed BatchNorm
    moments_kernel<<<64, 256, 0, stream>>>((const float2*)pY, pMom, N);
    // K2b: fold BN + pack/swizzle Wm bf16 + fold output projections
    prep_kernel<<<1, 512, 0, stream>>>(pMom, Wv, gamma, beta, Wm, Wnp, bnp,
                                       Wb_root, Wb_msg, bb, Wf_root, Wf_msg, bf_,
                                       Wpb, bpb, Wpf, bpf,
                                       pA0, pA1, pC, pWmBf, pCvec, N, HID);
    // K3: WMMA GEMM -> feat[N,32]
    feat_wmma_kernel<<<N / 128, 256, 0, stream>>>((const float2*)pY, pA0, pA1, pC,
                                                  pWmBf, bm, prelu, pFeat, N, HID);
    // K3b: per-node scalars
    node_vec_kernel<<<(N + 255) / 256, 256, 0, stream>>>(pFeat, pCvec, pBase, pSb, pSf, N);
    // K4a: inverse norms (+ optional bf16 mirror)
    norm_convert_kernel<<<(N + 7) / 8, 256, 0, stream>>>(vis_body, pInvb,
                                                         useBf ? pVbfB : nullptr, VB, N);
    norm_convert_kernel<<<(N + 7) / 8, 256, 0, stream>>>(vis_face, pInvf,
                                                         useBf ? pVbfF : nullptr, VF, N);
    // K4c: edge cosine + scalar message accumulation
    const int eblocks = (E + 7) / 8;
    if (useBf) {
        edge_cos_bf16_kernel<<<eblocks, 256, 0, stream>>>((const unsigned short*)pVbfB,
            pInvb, e_body, pSb, pNumb, pDegb, E, VB);
        edge_cos_bf16_kernel<<<eblocks, 256, 0, stream>>>((const unsigned short*)pVbfF,
            pInvf, e_face, pSf, pNumf, pDegf, E, VF);
    } else {
        edge_cos_f32_kernel<<<eblocks, 256, 0, stream>>>(vis_body, pInvb, e_body,
            pSb, pNumb, pDegb, E, VB);
        edge_cos_f32_kernel<<<eblocks, 256, 0, stream>>>(vis_face, pInvf, e_face,
            pSf, pNumf, pDegf, E, VF);
    }
    // K6: final combine
    final_kernel<<<(N + 255) / 256, 256, 0, stream>>>(pBase, pNumb, pDegb, pNumf, pDegf,
                                                      (float*)d_out, N);
}
